// BaseSelfAttention_57028575756702
// MI455X (gfx1250) — matrix operand order repaired and checked by reading, not compiled, not run on hardware
//
#include <hip/hip_runtime.h>
#include <hip/hip_bf16.h>

typedef __attribute__((ext_vector_type(16))) _Float16 v16h;
typedef __attribute__((ext_vector_type(8)))  _Float16 v8h;
typedef __attribute__((ext_vector_type(4)))  _Float16 v4h;
typedef __attribute__((ext_vector_type(8)))  float    v8f;

#define Dm  1024
#define Tm  2048
#define Bm  2
#define Hm  16
#define DHm 64
#define Mm  (Bm * Tm)   // 4096 rows total

static __device__ __forceinline__ v16h make16(v8h lo, v8h hi) {
  v16h r;
#pragma unroll
  for (int i = 0; i < 8; ++i) { r[i] = lo[i]; r[i + 8] = hi[i]; }
  return r;
}

static __device__ __forceinline__ v8f wmma16(v16h a, v16h b, v8f c) {
  // D = A(16x32 f16) * B(32x16 f16) + C(16x16 f32)
  return __builtin_amdgcn_wmma_f32_16x16x32_f16(
      /*neg_a=*/false, a, /*neg_b=*/false, b,
      /*c_mod=*/(short)0, c, /*reuse_a=*/false, /*reuse_b=*/false);
}

// CDNA5 async copy: global -> LDS, 32 contiguous bytes per lane.
// INST_OFFSET is added to BOTH the LDS and global addresses (ISA 15.18.3),
// so one base pair + offset:16 moves both 16B halves. Tracked by ASYNCcnt.
static __device__ __forceinline__ void async_copy32(unsigned lds_addr,
                                                    const void* gptr) {
  asm volatile(
      "global_load_async_to_lds_b128 %0, %1, off\n\t"
      "global_load_async_to_lds_b128 %0, %1, off offset:16"
      :
      : "v"(lds_addr), "v"((unsigned long long)gptr)
      : "memory");
}
// Async loads complete in issue order: "<=2 outstanding" == oldest tile landed.
static __device__ __forceinline__ void async_wait2() {
  asm volatile("s_wait_asynccnt 2" ::: "memory");
}
static __device__ __forceinline__ void async_wait0() {
  asm volatile("s_wait_asynccnt 0" ::: "memory");
}
static __device__ __forceinline__ unsigned ldsAddr(const void* p) {
  // generic LDS pointer: low 32 bits are the LDS byte offset
  return (unsigned)(unsigned long long)p;
}

static __device__ __forceinline__ float rmax16(float v) {
  v = fmaxf(v, __shfl_xor(v, 1, 16));
  v = fmaxf(v, __shfl_xor(v, 2, 16));
  v = fmaxf(v, __shfl_xor(v, 4, 16));
  v = fmaxf(v, __shfl_xor(v, 8, 16));
  return v;
}
static __device__ __forceinline__ float rsum16(float v) {
  v += __shfl_xor(v, 1, 16);
  v += __shfl_xor(v, 2, 16);
  v += __shfl_xor(v, 4, 16);
  v += __shfl_xor(v, 8, 16);
  return v;
}

// ---------------------------------------------------------------- fp32->fp16
__global__ __launch_bounds__(256) void cvt_kernel(const float* __restrict__ s,
                                                  _Float16* __restrict__ d) {
  int i = (blockIdx.x * 256 + threadIdx.x) * 4;   // grid sized exactly
  float4 v = *(const float4*)(s + i);
  v4h o;
  o[0] = (_Float16)v.x; o[1] = (_Float16)v.y;
  o[2] = (_Float16)v.z; o[3] = (_Float16)v.w;
  *(v4h*)(d + i) = o;
}

// ---------------------------------------------------------------- layernorm
__global__ __launch_bounds__(256) void ln_kernel(const float* __restrict__ x,
                                                 const float* __restrict__ g,
                                                 const float* __restrict__ b,
                                                 _Float16* __restrict__ xn) {
  int row = blockIdx.x;
  const float* xr = x + (size_t)row * Dm;
  float4 v = ((const float4*)xr)[threadIdx.x];          // 256 * 4 = 1024
  float s  = v.x + v.y + v.z + v.w;
  float s2 = v.x * v.x + v.y * v.y + v.z * v.z + v.w * v.w;
#pragma unroll
  for (int m = 16; m >= 1; m >>= 1) {
    s  += __shfl_xor(s,  m, 32);
    s2 += __shfl_xor(s2, m, 32);
  }
  __shared__ float ls[8], ls2[8];
  int w = threadIdx.x >> 5, l = threadIdx.x & 31;
  if (l == 0) { ls[w] = s; ls2[w] = s2; }
  __syncthreads();
  if (threadIdx.x == 0) {
    float a = 0.f, c = 0.f;
#pragma unroll
    for (int i = 0; i < 8; ++i) { a += ls[i]; c += ls2[i]; }
    ls[0] = a; ls2[0] = c;
  }
  __syncthreads();
  float mean = ls[0] * (1.0f / Dm);
  float var  = ls2[0] * (1.0f / Dm) - mean * mean;
  float rstd = rsqrtf(var + 1e-5f);
  int c0 = threadIdx.x * 4;
  float4 gg = ((const float4*)g)[threadIdx.x];
  float4 bb = ((const float4*)b)[threadIdx.x];
  v4h o;
  o[0] = (_Float16)((v.x - mean) * rstd * gg.x + bb.x);
  o[1] = (_Float16)((v.y - mean) * rstd * gg.y + bb.y);
  o[2] = (_Float16)((v.z - mean) * rstd * gg.z + bb.z);
  o[3] = (_Float16)((v.w - mean) * rstd * gg.w + bb.w);
  *(v4h*)(xn + (size_t)row * Dm + c0) = o;
}

// ------------------------------------------------- QKV projection (WMMA GEMM)
// q/k/v[t,n] = xn[t,:] . W[n,:] + bias[n]   (B column n == W row n, contiguous)
// Double-buffered async W staging; last tile peeled so the hot loop is
// branch-free (keeps WMMA accumulating in-place, no phi copies / hazard nops).
__global__ __launch_bounds__(128) void qkv_gemm(
    const _Float16* __restrict__ xn,
    const _Float16* wq, const _Float16* wk, const _Float16* wv,
    const float* bq, const float* bk, const float* bv,
    const float* __restrict__ mask,
    _Float16* oq, _Float16* ok, _Float16* ov) {
  int z = blockIdx.z;
  const _Float16* w   = (z == 0) ? wq : ((z == 1) ? wk : wv);
  const float*    bia = (z == 0) ? bq : ((z == 1) ? bk : bv);
  _Float16*       o   = (z == 0) ? oq : ((z == 1) ? ok : ov);

  __shared__ _Float16 wt[2][64][48];       // 48-half stride -> 32B aligned cols

  int tid = threadIdx.x;
  int wave = tid >> 5, lane = tid & 31;
  int r = lane & 15, gsel = lane >> 4;
  int n0 = blockIdx.x * 64;
  int row0 = blockIdx.y * 64 + wave * 16;

  int wn = tid >> 1, wc = (tid & 1) * 16;  // this thread's W-tile slice
  unsigned wt_lds[2] = { ldsAddr(&wt[0][wn][wc]), ldsAddr(&wt[1][wn][wc]) };
  const _Float16* wrow = w + (size_t)(n0 + wn) * Dm + wc;

  v8f acc[4];
#pragma unroll
  for (int j = 0; j < 4; ++j)
#pragma unroll
    for (int i = 0; i < 8; ++i) acc[j][i] = 0.f;

  auto compute = [&](int buf, int k0) {
    const _Float16* ap = xn + (size_t)(row0 + r) * Dm + k0 + gsel * 16;
    v16h a = make16(*(const v8h*)ap, *(const v8h*)(ap + 8));
#pragma unroll
    for (int j = 0; j < 4; ++j) {
      v16h bm = *(const v16h*)&wt[buf][j * 16 + r][gsel * 16];
      acc[j] = wmma16(a, bm, acc[j]);
    }
  };

  const int NT = Dm / 32;                  // 32 K-tiles
  async_copy32(wt_lds[0], wrow);           // prologue: tile 0 -> buf 0

  int t = 0;
  for (; t < NT - 1; ++t) {                // branch-free steady state
    int buf = t & 1;
    async_copy32(wt_lds[buf ^ 1], wrow + (t + 1) * 32);
    async_wait2();                         // tile t landed
    __syncthreads();
    compute(buf, t * 32);
    __syncthreads();                       // buf reusable for tile t+2
  }
  async_wait0();                           // peeled last tile
  __syncthreads();
  compute(t & 1, t * 32);

#pragma unroll
  for (int j = 0; j < 4; ++j) {
    int col = n0 + j * 16 + r;
    float bc = bia[col];
#pragma unroll
    for (int i = 0; i < 8; ++i) {
      int orow = row0 + i + 8 * gsel;
      float val = acc[j][i] + bc;
      if (z == 2) val *= mask[orow];                   // v is key-masked
      o[(size_t)orow * Dm + col] = (_Float16)val;
    }
  }
}

// ---------------------------------------------------- flash attention + resid
__global__ __launch_bounds__(128) void attn_kernel(
    const _Float16* __restrict__ q, const _Float16* __restrict__ k,
    const _Float16* __restrict__ v, const float* __restrict__ mask,
    const float* __restrict__ x, float* __restrict__ out) {
  __shared__ _Float16 kt[2][32][80];       // double-buffered K tile (row major)
  __shared__ _Float16 vt[64][48];          // transposed: 64 dims x 32 keys
  __shared__ _Float16 pl[4][16][48];       // per-wave P tile 16x32

  int b = blockIdx.z, h = blockIdx.y;
  int tid = threadIdx.x;
  int wave = tid >> 5, lane = tid & 31;
  int r = lane & 15, gsel = lane >> 4;
  size_t rowbase = (size_t)b * Tm;
  int n0 = blockIdx.x * 64 + wave * 16;    // this wave's 16 query rows
  int hoff = h * DHm;

  // Q tile in A-layout, two dh-chunks of 32
  v16h qa[2];
#pragma unroll
  for (int s = 0; s < 2; ++s) {
    const _Float16* p = q + (rowbase + n0 + r) * Dm + hoff + s * 32 + gsel * 16;
    qa[s] = make16(*(const v8h*)p, *(const v8h*)(p + 8));
  }

  int key = tid >> 2, dseg = (tid & 3) * 16;   // staging slice for this thread
  unsigned kt_lds[2] = { ldsAddr(&kt[0][key][dseg]), ldsAddr(&kt[1][key][dseg]) };
  const _Float16* kbase = k + (rowbase + key) * Dm + hoff + dseg;
  const _Float16* vbase = v + (rowbase + key) * Dm + hoff + dseg;

  float mi[8], li[8];
  v8f acc[4];
#pragma unroll
  for (int i = 0; i < 8; ++i) { mi[i] = -1e30f; li[i] = 0.f; }
#pragma unroll
  for (int j = 0; j < 4; ++j)
#pragma unroll
    for (int i = 0; i < 8; ++i) acc[j][i] = 0.f;

  auto compute = [&](int buf, int kk) {
    // --- S = Q K^T : two 16x16 tiles (keys 0-15, 16-31), K-dim = 64 (2 steps)
    v8f s0, s1;
#pragma unroll
    for (int i = 0; i < 8; ++i) { s0[i] = 0.f; s1[i] = 0.f; }
#pragma unroll
    for (int s = 0; s < 2; ++s) {
      int c = s * 32 + gsel * 16;
      v16h b0 = make16(*(const v8h*)&kt[buf][r][c],
                       *(const v8h*)&kt[buf][r][c + 8]);
      v16h b1 = make16(*(const v8h*)&kt[buf][16 + r][c],
                       *(const v8h*)&kt[buf][16 + r][c + 8]);
      s0 = wmma16(qa[s], b0, s0);
      s1 = wmma16(qa[s], b1, s1);
    }

    // --- key mask (column-uniform per lane in C-layout)
    float mk0 = (1.0f - mask[rowbase + kk + r])      * -1000000.0f;
    float mk1 = (1.0f - mask[rowbase + kk + 16 + r]) * -1000000.0f;
#pragma unroll
    for (int i = 0; i < 8; ++i) { s0[i] += mk0; s1[i] += mk1; }

    // --- online softmax (row stats via 16-lane shuffles)
    float p0[8], p1[8];
#pragma unroll
    for (int i = 0; i < 8; ++i) {
      float mx = rmax16(fmaxf(s0[i], s1[i]));
      float mnew = fmaxf(mi[i], mx);
      float scale = __expf(mi[i] - mnew);
      p0[i] = __expf(s0[i] - mnew);
      p1[i] = __expf(s1[i] - mnew);
      float rs = rsum16(p0[i] + p1[i]);
      li[i] = li[i] * scale + rs;
      mi[i] = mnew;
#pragma unroll
      for (int j = 0; j < 4; ++j) acc[j][i] *= scale;
    }

    // --- P: C-layout regs -> LDS -> A-layout regs (f16)
#pragma unroll
    for (int i = 0; i < 8; ++i) {
      pl[wave][i + 8 * gsel][r]      = (_Float16)p0[i];
      pl[wave][i + 8 * gsel][16 + r] = (_Float16)p1[i];
    }
    v16h pa = make16(*(const v8h*)&pl[wave][r][gsel * 16],
                     *(const v8h*)&pl[wave][r][gsel * 16 + 8]);

    // --- acc += P V  (K-dim = 32 keys, 4 output dim-tiles)
#pragma unroll
    for (int j = 0; j < 4; ++j) {
      v16h vb = *(const v16h*)&vt[j * 16 + r][gsel * 16];
      acc[j] = wmma16(pa, vb, acc[j]);
    }
  };

  const int NT = Tm / 32;                  // 64 KV tiles
  async_copy32(kt_lds[0], kbase);          // prologue: K tile 0
  v16h vv = *(const v16h*)vbase;           // prologue: V tile 0 in regs

  int t = 0;
  for (; t < NT - 1; ++t) {                // branch-free steady state
    int buf = t & 1;
    async_copy32(kt_lds[buf ^ 1], kbase + (size_t)(t + 1) * 32 * Dm);
    v16h vnext = *(const v16h*)(vbase + (size_t)(t + 1) * 32 * Dm);
    __builtin_prefetch(kbase + (size_t)(t + 2) * 32 * Dm, 0, 0);
    __builtin_prefetch(vbase + (size_t)(t + 2) * 32 * Dm, 0, 0);

#pragma unroll
    for (int i = 0; i < 16; ++i) vt[dseg + i][key] = vv[i];  // transpose V(t)

    async_wait2();                         // K(t) landed, K(t+1) in flight
    __syncthreads();
    compute(buf, t * 32);
    __syncthreads();                       // vt/pl reusable, kt[buf] for t+2
    vv = vnext;
  }
  // peeled last tile
#pragma unroll
  for (int i = 0; i < 16; ++i) vt[dseg + i][key] = vv[i];
  async_wait0();
  __syncthreads();
  compute(t & 1, t * 32);

  // --- normalize, residual add, store fp32
#pragma unroll
  for (int j = 0; j < 4; ++j) {
#pragma unroll
    for (int i = 0; i < 8; ++i) {
      int orow = n0 + i + 8 * gsel;
      int ocol = hoff + j * 16 + r;
      size_t idx = (rowbase + orow) * Dm + ocol;
      out[idx] = x[idx] + acc[j][i] / li[i];
    }
  }
}

// ---------------------------------------------------------------------- host
extern "C" void kernel_launch(void* const* d_in, const int* in_sizes, int n_in,
                              void* d_out, int out_size, void* d_ws, size_t ws_size,
                              hipStream_t stream) {
  const float* x    = (const float*)d_in[0];
  const float* mask = (const float*)d_in[1];
  const float* ln_g = (const float*)d_in[2];
  const float* ln_b = (const float*)d_in[3];
  const float* Wq   = (const float*)d_in[4];
  const float* bq   = (const float*)d_in[5];
  const float* Wk   = (const float*)d_in[6];
  const float* bk   = (const float*)d_in[7];
  const float* Wv   = (const float*)d_in[8];
  const float* bv   = (const float*)d_in[9];
  float* out = (float*)d_out;

  char* ws = (char*)d_ws;
  _Float16* xnh = (_Float16*)(ws);                          // 8 MiB
  _Float16* qh  = (_Float16*)(ws + ((size_t)8  << 20));     // 8 MiB
  _Float16* kh  = (_Float16*)(ws + ((size_t)16 << 20));     // 8 MiB
  _Float16* vh  = (_Float16*)(ws + ((size_t)24 << 20));     // 8 MiB
  _Float16* wqh = (_Float16*)(ws + ((size_t)32 << 20));     // 2 MiB each
  _Float16* wkh = (_Float16*)(ws + ((size_t)34 << 20));
  _Float16* wvh = (_Float16*)(ws + ((size_t)36 << 20));

  // weights fp32 -> fp16 (D*D each, 4 elems/thread)
  cvt_kernel<<<Dm * Dm / 1024, 256, 0, stream>>>(Wq, wqh);
  cvt_kernel<<<Dm * Dm / 1024, 256, 0, stream>>>(Wk, wkh);
  cvt_kernel<<<Dm * Dm / 1024, 256, 0, stream>>>(Wv, wvh);

  // layernorm -> xn (fp16)
  ln_kernel<<<Mm, 256, 0, stream>>>(x, ln_g, ln_b, xnh);

  // q/k/v = xn @ W^T + b  (64x64 tiles, 4 waves/block, z selects matrix)
  qkv_gemm<<<dim3(Dm / 64, Mm / 64, 3), 128, 0, stream>>>(
      xnh, wqh, wkh, wvh, bq, bk, bv, mask, qh, kh, vh);

  // flash attention + residual
  attn_kernel<<<dim3(Tm / 64, Hm, Bm), 128, 0, stream>>>(
      qh, kh, vh, mask, x, out);
}
